// Decoder_zip_63986422776327
// MI455X (gfx1250) — matrix operand order repaired and checked by reading, not compiled, not run on hardware
//
#include <hip/hip_runtime.h>
#include <math.h>

typedef __attribute__((ext_vector_type(16))) _Float16 v16h;
typedef __attribute__((ext_vector_type(8)))  float    v8f;
typedef __attribute__((ext_vector_type(4)))  float    v4f;

#define N_ROWS 65536
#define DZ     128
#define DQ     512
#define M_PER_WAVE 16              // mTiles processed by each wave

__device__ __forceinline__ unsigned pcg_hash(unsigned x) {
    x = x * 747796405u + 2891336453u;
    unsigned w = ((x >> ((x >> 28u) + 4u)) ^ x) * 277803737u;
    return (w >> 22u) ^ w;
}

__device__ __forceinline__ float u01(unsigned h) {
    // (0,1) uniform from 24 high-entropy bits
    return (float)(h >> 8) * (1.0f / 16777216.0f) + 1e-7f;
}

__device__ __forceinline__ v16h load_cvt_b16(const float* p) {
    // 16 contiguous f32 -> 16 f16 (compiler emits v_cvt_pk_f16_f32 pairs)
    v16h r;
#pragma unroll
    for (int c = 0; c < 4; ++c) {
        v4f t = *(const v4f*)(p + 4 * c);
#pragma unroll
        for (int j = 0; j < 4; ++j) r[4 * c + j] = (_Float16)t[j];
    }
    return r;
}

__global__ __launch_bounds__(256) void zip_decoder_kernel(
    const float* __restrict__ z,
    const float* __restrict__ beta,
    const float* __restrict__ gamma,
    float* __restrict__ out)
{
    const int wave = threadIdx.x >> 5;
    const int lane = threadIdx.x & 31;
    const int w    = blockIdx.x * 8 + wave;       // 8192 waves total
    const int qTile  = w & 31;                    // 32 tiles across DQ
    const int mChunk = w >> 5;                    // 256 chunks of 16 mTiles
    const int half   = lane >> 4;                 // 0: lanes 0-15, 1: lanes 16-31
    const int lr     = lane & 15;

    const int qCol = qTile * 16 + lr;             // B-matrix column owned by this lane
    const int q    = qCol;                        // global output column

    // ---- Hoist B fragments (32x16 f16 each k-step) into registers: reused
    // across all 16 mTiles. Lane holds column q, K = ks*32 + half*16 + [0..15],
    // contiguous in row-major beta/gamma [q][d].
    v16h bb[4], bg[4];
#pragma unroll
    for (int ks = 0; ks < 4; ++ks) {
        const int koff = ks * 32 + half * 16;
        bb[ks] = load_cvt_b16(beta  + (size_t)qCol * DZ + koff);
        bg[ks] = load_cvt_b16(gamma + (size_t)qCol * DZ + koff);
    }

    for (int it = 0; it < M_PER_WAVE; ++it) {
        const int mTile = mChunk * M_PER_WAVE + it;
        const int aRow  = mTile * 16 + lr;        // A-matrix row owned by this lane

        v8f accB = {};                            // logits accumulator
        v8f accG = {};                            // log-rate accumulator

#pragma unroll
        for (int ks = 0; ks < 4; ++ks) {          // K = 128 in 4 steps of 32
            // A fragment (16x32 f16), contiguous K order matching B:
            // lanes 0-15: K = ks*32 + [0..15]
            // lanes 16-31: K = ks*32 + [16..31]
            v16h a;
            {
                const float* pz = z + (size_t)aRow * DZ + ks * 32 + half * 16;
                v4f z0 = *(const v4f*)(pz + 0);
                v4f z1 = *(const v4f*)(pz + 4);
                v4f z2 = *(const v4f*)(pz + 8);
                v4f z3 = *(const v4f*)(pz + 12);
#pragma unroll
                for (int j = 0; j < 4; ++j) {
                    a[j]      = (_Float16)z0[j];
                    a[4 + j]  = (_Float16)z1[j];
                    a[8 + j]  = (_Float16)z2[j];
                    a[12 + j] = (_Float16)z3[j];
                }
            }
            accB = __builtin_amdgcn_wmma_f32_16x16x32_f16(
                false, a, false, bb[ks], (short)0, accB, false, false);
            accG = __builtin_amdgcn_wmma_f32_16x16x32_f16(
                false, a, false, bg[ks], (short)0, accG, false, false);
        }

        // ---- Fused elementwise: C/D layout: VGPR i holds (M = i + half*8, N = lr)
#pragma unroll
        for (int i = 0; i < 8; ++i) {
            const int m = i + half * 8;
            const int n = mTile * 16 + m;         // global output row
            const unsigned idx = (unsigned)(n * DQ + q);

            float logit = accB[i];
            float lgr   = accG[i];

            // Bernoulli(1 - pi) mask without any division:
            //   u < 1 - sigmoid(logit) = 1/(1 + e^logit)  <=>  fma(u, e^logit, u) < 1
            // (e^logit overflow -> inf -> keep=false, matching pi -> 1)
            float e1 = __expf(logit);
            unsigned hb = pcg_hash(idx + 0x85EBCA6Bu);
            float u2 = u01(hb);
            bool keep = (fmaf(u2, e1, u2) < 1.0f);

            float result = 0.0f;
            if (keep) {
                // rate = nan_to_num(exp(.)); |rate| + 0.01; clamp for sampler safety
                float rate = __expf(lgr);
                if (!(rate == rate)) rate = 1e-4f;
                rate = fabsf(rate) + 0.01f;
                if (rate > 60.0f) rate = 60.0f;

                // Knuth Poisson: multiply uniforms until product <= exp(-rate)
                float L = __expf(-rate);
                unsigned s = pcg_hash(idx ^ 0x9E3779B9u);
                float p = 1.0f;
                int k = -1;
                while (p > L && k < 200) {
                    s = s * 1664525u + 1013904223u;
                    p *= u01(s ^ (s >> 16));
                    ++k;
                }
                result = (float)k;
            }
            out[(size_t)n * DQ + q] = result;
        }
    }
}

extern "C" void kernel_launch(void* const* d_in, const int* in_sizes, int n_in,
                              void* d_out, int out_size, void* d_ws, size_t ws_size,
                              hipStream_t stream) {
    const float* z     = (const float*)d_in[0];
    const float* beta  = (const float*)d_in[1];
    const float* gamma = (const float*)d_in[2];
    float* out = (float*)d_out;

    // 32 qTiles x 256 mChunks = 8192 waves, 8 waves (256 threads) per block
    const int nBlocks = (DQ / 16) * (N_ROWS / 16 / M_PER_WAVE) / 8;   // 1024
    zip_decoder_kernel<<<nBlocks, 256, 0, stream>>>(z, beta, gamma, out);
}
